// Encoder_8323646619980
// MI455X (gfx1250) — compile-verified, matrix operand order checked by reading, not run on hardware
//
#include <hip/hip_runtime.h>
#include <hip/hip_bf16.h>
#include <cmath>

#define N_NODES  20000
#define N_EDGES  320000
#define ET       (N_EDGES + N_NODES)   // self-loops appended
#define F_IN     9
#define EMB      10
#define HID      64
#define HEADS    10
#define HH       (HEADS * HID)         // 640
#define OUT_D    512
#define EDGE_F   6
#define N_GRAPHS 512
#define DIMF     266                   // 10 + 4*64
#define EPS      1e-5f

typedef __attribute__((ext_vector_type(16))) _Float16 v16h;
typedef __attribute__((ext_vector_type(8)))  _Float16 v8h;
typedef __attribute__((ext_vector_type(4)))  _Float16 v4h;
typedef __attribute__((ext_vector_type(8)))  float    v8f;

// ---------- order-preserving float<->uint for atomic max ----------
__device__ __forceinline__ unsigned fenc(float f) {
    unsigned u = __float_as_uint(f);
    return (u & 0x80000000u) ? ~u : (u | 0x80000000u);
}
__device__ __forceinline__ float fdec(unsigned u) {
    unsigned b = (u & 0x80000000u) ? (u & 0x7FFFFFFFu) : ~u;
    return __uint_as_float(b);
}

// ---------- generic fill ----------
__global__ void fill_u32_kernel(unsigned* __restrict__ p, unsigned v, long n) {
    long i = (long)blockIdx.x * blockDim.x + threadIdx.x;
    long stride = (long)gridDim.x * blockDim.x;
    for (; i < n; i += stride) p[i] = v;
}

// ---------- pack f32 -> padded f16 [rowsP][Kp] (zero-pad ragged edges) ----------
__global__ void pack_f16_kernel(const float* __restrict__ src, int ld_src,
                                int rows, int cols,
                                _Float16* __restrict__ dst, int Kp, long total) {
    long i = (long)blockIdx.x * blockDim.x + threadIdx.x;
    if (i >= total) return;
    int r = (int)(i / Kp), c = (int)(i % Kp);
    float v = 0.0f;
    if (r < rows && c < cols) v = src[(size_t)r * ld_src + c];
    dst[i] = (_Float16)v;
}

// ---------- WMMA GEMM on pre-packed f16: C[M,N] = A * B^T (+bias) ----------
// A: [M][Kp] f16 row-major (M % 32 == 0), B: [Np][Kp] f16 (Np % 128 == 0, Kp % 32 == 0).
// Block tile 32(M) x 128(N); 8 waves; each wave: one A fragment feeds two WMMAs.
// Double-buffered LDS, register prefetch, one barrier per K-step, no bounds
// logic and no conversions in the hot loop.
__global__ void __launch_bounds__(256)
gemm_wmma_f16_kernel(const _Float16* __restrict__ A,
                     const _Float16* __restrict__ B,
                     const float* __restrict__ bias,
                     float* __restrict__ C, int ldc,
                     int M, int N, int Kp)
{
    __shared__ alignas(16) _Float16 As[2][32][40];    // [buf][m][k], 80B row stride
    __shared__ alignas(16) _Float16 Bs[2][128][40];   // [buf][n][k]
    const int tid  = threadIdx.x;
    const int lane = tid & 31;
    const int wave = tid >> 5;
    const int wm   = wave >> 2;        // 0..1
    const int wn   = wave & 3;         // 0..3
    const int m0   = blockIdx.y * 32;
    const int n0   = blockIdx.x * 128;

    // staging assignments (fixed per thread)
    const int ar = tid >> 3, ac = (tid & 7) * 4;            // A: 4 halfs/thread
    const int j0  = tid * 2;
    const int br0 = j0 >> 2,       bc0 = (j0 & 3) * 8;      // B: 2x 8 halfs/thread
    const int br1 = (j0 + 1) >> 2, bc1 = ((j0 + 1) & 3) * 8;

    const _Float16* Ap  = A + (size_t)(m0 + ar)  * Kp + ac;
    const _Float16* Bp0 = B + (size_t)(n0 + br0) * Kp + bc0;
    const _Float16* Bp1 = B + (size_t)(n0 + br1) * Kp + bc1;

    v8f acc0 = {}, acc1 = {};
    const int row = lane & 15;
    const int khi = (lane >> 4) << 3;  // lanes 0-15 -> 0, lanes 16-31 -> 8

    const int nk = Kp >> 5;

    // prologue: tile 0
    v4h ra  = *(const v4h*)Ap;
    v8h rb0 = *(const v8h*)Bp0;
    v8h rb1 = *(const v8h*)Bp1;
    *(v4h*)&As[0][ar][ac]   = ra;
    *(v8h*)&Bs[0][br0][bc0] = rb0;
    *(v8h*)&Bs[0][br1][bc1] = rb1;
    __syncthreads();

    for (int t = 0; t < nk; ++t) {
        const int cur = t & 1, nxt = cur ^ 1;
        if (t + 1 < nk) {                 // prefetch next tile into registers
            const int ko = (t + 1) << 5;
            ra  = *(const v4h*)(Ap  + ko);
            rb0 = *(const v8h*)(Bp0 + ko);
            rb1 = *(const v8h*)(Bp1 + ko);
        }
        // fragments: two ds_load_b128 each (half h holds k 8h..8h+7 then 16+8h..16+8h+7, AMD table order)
        v8h alo = *(const v8h*)&As[cur][wm * 16 + row][khi];
        v8h ahi = *(const v8h*)&As[cur][wm * 16 + row][16 + khi];
        v16h af = __builtin_shufflevector(alo, ahi,
                   0,1,2,3,4,5,6,7,8,9,10,11,12,13,14,15);
        v8h b0l = *(const v8h*)&Bs[cur][wn * 32 + row][khi];
        v8h b0h = *(const v8h*)&Bs[cur][wn * 32 + row][16 + khi];
        v16h bf0 = __builtin_shufflevector(b0l, b0h,
                   0,1,2,3,4,5,6,7,8,9,10,11,12,13,14,15);
        v8h b1l = *(const v8h*)&Bs[cur][wn * 32 + 16 + row][khi];
        v8h b1h = *(const v8h*)&Bs[cur][wn * 32 + 16 + row][16 + khi];
        v16h bf1 = __builtin_shufflevector(b1l, b1h,
                   0,1,2,3,4,5,6,7,8,9,10,11,12,13,14,15);

        acc0 = __builtin_amdgcn_wmma_f32_16x16x32_f16(
            false, af, false, bf0, (short)0, acc0, false, false);
        acc1 = __builtin_amdgcn_wmma_f32_16x16x32_f16(
            false, af, false, bf1, (short)0, acc1, false, false);

        if (t + 1 < nk) {                 // stage next tile into other buffer
            *(v4h*)&As[nxt][ar][ac]   = ra;
            *(v8h*)&Bs[nxt][br0][bc0] = rb0;
            *(v8h*)&Bs[nxt][br1][bc1] = rb1;
            __syncthreads();
        }
    }

    // epilogue: f32 C/D layout (lane<16: M=r; lane>=16: M=8+r; N=lane%16)
    const int ncol0 = n0 + wn * 32 + (lane & 15);
    const int ncol1 = ncol0 + 16;
    const int mbase = m0 + wm * 16 + ((lane >> 4) << 3);
    float bv0 = (bias != nullptr && ncol0 < N) ? bias[ncol0] : 0.0f;
    float bv1 = (bias != nullptr && ncol1 < N) ? bias[ncol1] : 0.0f;
#pragma unroll
    for (int r = 0; r < 8; ++r) {
        int mrow = mbase + r;
        if (mrow < M) {
            if (ncol0 < N) C[(size_t)mrow * ldc + ncol0] = acc0[r] + bv0;
            if (ncol1 < N) C[(size_t)mrow * ldc + ncol1] = acc1[r] + bv1;
        }
    }
}

// ---------- edge_attr column sums (for self-loop fill_value='mean') ----------
__global__ void edge_mean_kernel(const float* __restrict__ ea, float* __restrict__ sum) {
    __shared__ float sh[EDGE_F];
    if (threadIdx.x < EDGE_F) sh[threadIdx.x] = 0.0f;
    __syncthreads();
    float loc[EDGE_F] = {};
    for (int t = blockIdx.x * blockDim.x + threadIdx.x; t < N_EDGES;
         t += gridDim.x * blockDim.x)
        for (int f = 0; f < EDGE_F; ++f) loc[f] += ea[(size_t)t * EDGE_F + f];
    for (int f = 0; f < EDGE_F; ++f) atomicAdd(&sh[f], loc[f]);
    __syncthreads();
    if (threadIdx.x < EDGE_F) atomicAdd(&sum[threadIdx.x], sh[threadIdx.x]);
}

// ---------- fold lin_edge_W + att_edge into 6x10 w_e ----------
__global__ void we_kernel(const float* __restrict__ lin_edge_W,
                          const float* __restrict__ att_edge,
                          float* __restrict__ w_e) {
    int i = threadIdx.x;
    if (i >= EDGE_F * HEADS) return;
    int f = i % EDGE_F, h = i / EDGE_F;
    float s = 0.0f;
    for (int c = 0; c < HID; ++c)
        s += lin_edge_W[(size_t)(h * HID + c) * EDGE_F + f] * att_edge[h * HID + c];
    w_e[f * HEADS + h] = s;
}

// ---------- per-node per-head attention scores ----------
__global__ void att_score_kernel(const float* __restrict__ hs,
                                 const float* __restrict__ att_src,
                                 const float* __restrict__ att_dst,
                                 float* __restrict__ a_src,
                                 float* __restrict__ a_dst) {
    int i = blockIdx.x * blockDim.x + threadIdx.x;
    if (i >= N_NODES * HEADS) return;
    int n = i / HEADS, h = i % HEADS;
    const float4* hp = (const float4*)(hs + (size_t)n * HH + h * HID);
    const float4* as = (const float4*)(att_src + h * HID);
    const float4* ad = (const float4*)(att_dst + h * HID);
    float s1 = 0.0f, s2 = 0.0f;
#pragma unroll 4
    for (int q = 0; q < HID / 4; ++q) {
        float4 v = hp[q], a = as[q], b = ad[q];
        s1 += v.x * a.x + v.y * a.y + v.z * a.z + v.w * a.w;
        s2 += v.x * b.x + v.y * b.y + v.z * b.z + v.w * b.w;
    }
    a_src[i] = s1; a_dst[i] = s2;
}

// ---------- per-edge alpha + segment max ----------
__global__ void edge_alpha_kernel(const int* __restrict__ ei,
                                  const float* __restrict__ ea,
                                  const float* __restrict__ ea_sum,
                                  const float* __restrict__ w_e,
                                  const float* __restrict__ a_src,
                                  const float* __restrict__ a_dst,
                                  float* __restrict__ alpha,
                                  unsigned* __restrict__ amax) {
    int t = blockIdx.x * blockDim.x + threadIdx.x;
    if (t >= ET) return;
    int s, d;
    float eav[EDGE_F];
    if (t < N_EDGES) {
        s = ei[t]; d = ei[N_EDGES + t];
        for (int f = 0; f < EDGE_F; ++f) eav[f] = ea[(size_t)t * EDGE_F + f];
    } else {
        s = d = t - N_EDGES;
        const float inv = 1.0f / (float)N_EDGES;
        for (int f = 0; f < EDGE_F; ++f) eav[f] = ea_sum[f] * inv;
    }
#pragma unroll
    for (int h = 0; h < HEADS; ++h) {
        float aeh = 0.0f;
        for (int f = 0; f < EDGE_F; ++f) aeh += eav[f] * w_e[f * HEADS + h];
        float a = a_src[s * HEADS + h] + a_dst[d * HEADS + h] + aeh;
        a = (a > 0.0f) ? a : 0.2f * a;          // leaky_relu(0.2)
        alpha[(size_t)t * HEADS + h] = a;
        atomicMax(&amax[d * HEADS + h], fenc(a));
    }
}

// ---------- exp(alpha - max) + segment sum ----------
__global__ void edge_exp_kernel(const int* __restrict__ ei,
                                float* __restrict__ alpha,
                                const unsigned* __restrict__ amax,
                                float* __restrict__ den) {
    int t = blockIdx.x * blockDim.x + threadIdx.x;
    if (t >= ET) return;
    int d = (t < N_EDGES) ? ei[N_EDGES + t] : (t - N_EDGES);
#pragma unroll
    for (int h = 0; h < HEADS; ++h) {
        float m = fdec(amax[d * HEADS + h]);
        float e = expf(alpha[(size_t)t * HEADS + h] - m);
        alpha[(size_t)t * HEADS + h] = e;
        atomicAdd(&den[d * HEADS + h], e);
    }
}

// ---------- weighted message scatter: out[d] += coef * hs[s] ----------
__global__ void scatter_kernel(const int* __restrict__ ei,
                               const float* __restrict__ hs,
                               const float* __restrict__ ex,
                               const float* __restrict__ den,
                               float* __restrict__ outa) {
    long w = (long)blockIdx.x * blockDim.x + threadIdx.x;
    if (w >= (long)ET * 160) return;
    int t = (int)(w / 160);
    int j = (int)(w % 160);
    int h = j >> 4;
    int c = (j & 15) << 2;
    int s, d;
    if (t < N_EDGES) { s = ei[t]; d = ei[N_EDGES + t]; }
    else             { s = d = t - N_EDGES; }
    float coef = ex[(size_t)t * HEADS + h] / (den[d * HEADS + h] + 1e-16f);
    float4 v = *(const float4*)(hs + (size_t)s * HH + h * HID + c);
    float* o = outa + (size_t)d * HH + h * HID + c;
    atomicAdd(o + 0, v.x * coef);
    atomicAdd(o + 1, v.y * coef);
    atomicAdd(o + 2, v.z * coef);
    atomicAdd(o + 3, v.w * coef);
}

// ---------- head mean + conv bias ----------
__global__ void head_mean_kernel(const float* __restrict__ outa,
                                 const float* __restrict__ bias,
                                 float* __restrict__ xc) {
    int i = blockIdx.x * blockDim.x + threadIdx.x;
    if (i >= N_NODES * HID) return;
    int n = i / HID, c = i % HID;
    float s = 0.0f;
#pragma unroll
    for (int h = 0; h < HEADS; ++h) s += outa[(size_t)n * HH + h * HID + c];
    xc[i] = s * (1.0f / (float)HEADS) + bias[c];
}

// ---------- BN stats: per-column sum / sumsq (blockDim == C) ----------
__global__ void bn_stats_kernel(const float* __restrict__ x, int ld, int C, int Mrows,
                                float* __restrict__ sum, float* __restrict__ sumsq) {
    int c = threadIdx.x;
    if (c >= C) return;
    float s = 0.0f, s2 = 0.0f;
    for (int n = blockIdx.x; n < Mrows; n += gridDim.x) {
        float v = x[(size_t)n * ld + c];
        s += v; s2 += v * v;
    }
    atomicAdd(&sum[c], s);
    atomicAdd(&sumsq[c], s2);
}

// ---------- BN (training-mode biased stats) + exact GELU ----------
__global__ void bn_gelu_kernel(const float* __restrict__ x, int ldx,
                               float* __restrict__ y, int ldy, int yoff,
                               int C, int Mrows,
                               const float* __restrict__ sum,
                               const float* __restrict__ sumsq,
                               const float* __restrict__ g,
                               const float* __restrict__ b) {
    long i = (long)blockIdx.x * blockDim.x + threadIdx.x;
    if (i >= (long)Mrows * C) return;
    int n = (int)(i / C), c = (int)(i % C);
    float mu  = sum[c] / (float)Mrows;
    float var = sumsq[c] / (float)Mrows - mu * mu;
    float v = (x[(size_t)n * ldx + c] - mu) * rsqrtf(var + EPS) * g[c] + b[c];
    v = 0.5f * v * (1.0f + erff(v * 0.70710678118654752f));
    y[(size_t)n * ldy + yoff + c] = v;
}

// ---------- graph pooling ----------
__global__ void pool_count_kernel(const int* __restrict__ batch, float* __restrict__ cnt) {
    int n = blockIdx.x * blockDim.x + threadIdx.x;
    if (n < N_NODES) atomicAdd(&cnt[batch[n]], 1.0f);
}
__global__ void pool_sum_kernel(const int* __restrict__ batch,
                                const float* __restrict__ h3,
                                float* __restrict__ out) {
    long i = (long)blockIdx.x * blockDim.x + threadIdx.x;
    if (i >= (long)N_NODES * OUT_D) return;
    int n = (int)(i / OUT_D), c = (int)(i % OUT_D);
    atomicAdd(&out[(size_t)batch[n] * OUT_D + c], h3[i]);
}
__global__ void pool_div_kernel(float* __restrict__ out, const float* __restrict__ cnt) {
    int i = blockIdx.x * blockDim.x + threadIdx.x;
    if (i < N_GRAPHS * OUT_D) out[i] /= fmaxf(cnt[i / OUT_D], 1.0f);
}

// =================================================================
extern "C" void kernel_launch(void* const* d_in, const int* in_sizes, int n_in,
                              void* d_out, int out_size, void* d_ws, size_t ws_size,
                              hipStream_t stream) {
    const float* x     = (const float*)d_in[0];
    const int*   ei    = (const int*)  d_in[1];
    const float* ea    = (const float*)d_in[2];
    const int*   batch = (const int*)  d_in[3];
    const float* emb_W = (const float*)d_in[4];
    const float* emb_b = (const float*)d_in[5];
    const float* lin1_W    = (const float*)d_in[38];
    const float* lin1_b    = (const float*)d_in[39];
    const float* bn1_gamma = (const float*)d_in[40];
    const float* bn1_beta  = (const float*)d_in[41];
    const float* lin2_W    = (const float*)d_in[42];
    const float* lin2_b    = (const float*)d_in[43];
    float* out = (float*)d_out;

    // ---- workspace carve (256B aligned) ----
    char* ws = (char*)d_ws;
    size_t off = 0;
    auto alloc = [&](size_t bytes) -> void* {
        void* p = ws + off;
        off += (bytes + 255) & ~(size_t)255;
        return p;
    };
    float*     h      = (float*)    alloc((size_t)N_NODES * DIMF * 4);
    float*     hs     = (float*)    alloc((size_t)N_NODES * HH * 4);
    float*     outa   = (float*)    alloc((size_t)N_NODES * HH * 4);
    float*     a_src  = (float*)    alloc((size_t)N_NODES * HEADS * 4);
    float*     a_dst  = (float*)    alloc((size_t)N_NODES * HEADS * 4);
    float*     alpha  = (float*)    alloc((size_t)ET * HEADS * 4);
    unsigned*  amax   = (unsigned*) alloc((size_t)N_NODES * HEADS * 4);
    float*     den    = (float*)    alloc((size_t)N_NODES * HEADS * 4);
    float*     xc     = (float*)    alloc((size_t)N_NODES * HID * 4);
    float*     h2     = (float*)    alloc((size_t)N_NODES * OUT_D * 4);
    float*     h3     = (float*)    alloc((size_t)N_NODES * OUT_D * 4);
    _Float16*  Apk    = (_Float16*) alloc((size_t)N_NODES * 512 * 2);  // max Kp = 512
    _Float16*  Wpk    = (_Float16*) alloc((size_t)640 * 512 * 2);      // max Np*Kp
    float*     ea_sum = (float*)    alloc(EDGE_F * 4);
    float*     w_e    = (float*)    alloc(EDGE_F * HEADS * 4);
    float*     bnsum  = (float*)    alloc(OUT_D * 4);
    float*     bnsq   = (float*)    alloc(OUT_D * 4);
    float*     cnt    = (float*)    alloc(N_GRAPHS * 4);
    (void)ws_size; (void)in_sizes; (void)n_in; (void)out_size;

    auto fill0 = [&](void* p, long n_u32) {
        int blocks = (int)((n_u32 + 255) / 256);
        if (blocks > 65535) blocks = 65535;
        hipLaunchKernelGGL(fill_u32_kernel, dim3(blocks), dim3(256), 0, stream,
                           (unsigned*)p, 0u, n_u32);
    };
    // pack A and W to padded f16, then WMMA GEMM
    auto gemm = [&](const float* A, int lda, const float* W, int ldw,
                    const float* bias, float* C, int ldc, int M, int N, int K) {
        const int Kp = (K + 31) & ~31;
        const int Np = (N + 127) & ~127;
        const long ta = (long)M * Kp;
        hipLaunchKernelGGL(pack_f16_kernel,
                           dim3((unsigned)((ta + 255) / 256)), dim3(256), 0, stream,
                           A, lda, M, K, Apk, Kp, ta);
        const long tw = (long)Np * Kp;
        hipLaunchKernelGGL(pack_f16_kernel,
                           dim3((unsigned)((tw + 255) / 256)), dim3(256), 0, stream,
                           W, ldw, N, K, Wpk, Kp, tw);
        dim3 grid(Np / 128, (M + 31) / 32);
        hipLaunchKernelGGL(gemm_wmma_f16_kernel, grid, dim3(256), 0, stream,
                           Apk, Wpk, bias, C, ldc, M, N, Kp);
    };

    // ---- edge_attr mean (for self-loop fill) ----
    fill0(ea_sum, EDGE_F);
    hipLaunchKernelGGL(edge_mean_kernel, dim3(256), dim3(256), 0, stream, ea, ea_sum);

    // ---- node embedding: h[:,0:10] = x @ emb_W^T + emb_b ----
    gemm(x, F_IN, emb_W, F_IN, emb_b, h, DIMF, N_NODES, EMB, F_IN);

    const int dims[4] = {10, 74, 138, 202};
    for (int l = 0; l < 4; ++l) {
        const float* lin_W      = (const float*)d_in[6 + l * 8 + 0];
        const float* att_src_p  = (const float*)d_in[6 + l * 8 + 1];
        const float* att_dst_p  = (const float*)d_in[6 + l * 8 + 2];
        const float* lin_edge_W = (const float*)d_in[6 + l * 8 + 3];
        const float* att_edge   = (const float*)d_in[6 + l * 8 + 4];
        const float* conv_bias  = (const float*)d_in[6 + l * 8 + 5];
        const float* bn_gamma   = (const float*)d_in[6 + l * 8 + 6];
        const float* bn_beta    = (const float*)d_in[6 + l * 8 + 7];
        const int K = dims[l];

        fill0(amax, (long)N_NODES * HEADS);
        fill0(den,  (long)N_NODES * HEADS);
        fill0(outa, (long)N_NODES * HH);

        // hs = h[:, :K] @ lin_W^T   (WMMA)
        gemm(h, DIMF, lin_W, K, nullptr, hs, HH, N_NODES, HH, K);

        hipLaunchKernelGGL(att_score_kernel,
                           dim3((N_NODES * HEADS + 255) / 256), dim3(256), 0, stream,
                           hs, att_src_p, att_dst_p, a_src, a_dst);
        hipLaunchKernelGGL(we_kernel, dim3(1), dim3(64), 0, stream,
                           lin_edge_W, att_edge, w_e);
        hipLaunchKernelGGL(edge_alpha_kernel,
                           dim3((ET + 255) / 256), dim3(256), 0, stream,
                           ei, ea, ea_sum, w_e, a_src, a_dst, alpha, amax);
        hipLaunchKernelGGL(edge_exp_kernel,
                           dim3((ET + 255) / 256), dim3(256), 0, stream,
                           ei, alpha, amax, den);
        {
            long work = (long)ET * 160;
            hipLaunchKernelGGL(scatter_kernel,
                               dim3((unsigned)((work + 255) / 256)), dim3(256), 0, stream,
                               ei, hs, alpha, den, outa);
        }
        hipLaunchKernelGGL(head_mean_kernel,
                           dim3((N_NODES * HID + 255) / 256), dim3(256), 0, stream,
                           outa, conv_bias, xc);

        fill0(bnsum, OUT_D); fill0(bnsq, OUT_D);
        hipLaunchKernelGGL(bn_stats_kernel, dim3(128), dim3(HID), 0, stream,
                           xc, HID, HID, N_NODES, bnsum, bnsq);
        // BN + GELU, write into concat slot h[:, dims[l] : dims[l]+64]
        hipLaunchKernelGGL(bn_gelu_kernel,
                           dim3(((long)N_NODES * HID + 255) / 256), dim3(256), 0, stream,
                           xc, HID, h, DIMF, dims[l], HID, N_NODES,
                           bnsum, bnsq, bn_gamma, bn_beta);
    }

    // ---- lin1 + BN + GELU ----
    gemm(h, DIMF, lin1_W, DIMF, lin1_b, h2, OUT_D, N_NODES, OUT_D, DIMF);
    fill0(bnsum, OUT_D); fill0(bnsq, OUT_D);
    hipLaunchKernelGGL(bn_stats_kernel, dim3(128), dim3(OUT_D), 0, stream,
                       h2, OUT_D, OUT_D, N_NODES, bnsum, bnsq);
    hipLaunchKernelGGL(bn_gelu_kernel,
                       dim3(((long)N_NODES * OUT_D + 255) / 256), dim3(256), 0, stream,
                       h2, OUT_D, h2, OUT_D, 0, OUT_D, N_NODES,
                       bnsum, bnsq, bn1_gamma, bn1_beta);

    // ---- lin2 ----
    gemm(h2, OUT_D, lin2_W, OUT_D, lin2_b, h3, OUT_D, N_NODES, OUT_D, OUT_D);

    // ---- graph mean pooling ----
    fill0(out, (long)N_GRAPHS * OUT_D);
    fill0(cnt, N_GRAPHS);
    hipLaunchKernelGGL(pool_count_kernel,
                       dim3((N_NODES + 255) / 256), dim3(256), 0, stream, batch, cnt);
    hipLaunchKernelGGL(pool_sum_kernel,
                       dim3(((long)N_NODES * OUT_D + 255) / 256), dim3(256), 0, stream,
                       batch, h3, out);
    hipLaunchKernelGGL(pool_div_kernel,
                       dim3((N_GRAPHS * OUT_D + 255) / 256), dim3(256), 0, stream,
                       out, cnt);
}
